// PVCNN_11673721111025
// MI455X (gfx1250) — matrix operand order repaired and checked by reading, not compiled, not run on hardware
//
#include <hip/hip_runtime.h>
#include <math.h>

// ---------------------------------------------------------------------------
// PVCNN forward for MI455X (gfx1250, wave32, WMMA, double-buffered
// async-to-LDS weight staging).
// Input flattening assumption (jax pytree order, keys sorted):
//   0: inputs (4,6,8192)
//   cls:  1 b1, 2 b2, 3 b3, 4 bn1.beta, 5 bn1.gamma, 6 bn1.mean, 7 bn1.var,
//         8 bn2.beta, 9 bn2.gamma, 10 bn2.mean, 11 bn2.var, 12 w1, 13 w2, 14 w3
//   mlp4: 15 b, 16 bn.beta, 17 bn.gamma, 18 bn.mean, 19 bn.var, 20 w
//   mlp5: 21 b, 22 bn.beta, 23 bn.gamma, 24 bn.mean, 25 bn.var, 26 w
//   pvX (base = 27/49/71): +0..3 bn1.{beta,gamma,mean,var}, +4..7 bn2.{...},
//         +8 cb1, +9 cb2, +10 cw1, +11 cw2, +12 mlp.b, +13..16 mlp.bn.{...},
//         +17 mlp.w, +18 se.b1, +19 se.b2, +20 se.w1, +21 se.w2
// ---------------------------------------------------------------------------

typedef __attribute__((ext_vector_type(16))) _Float16 v16h;
typedef __attribute__((ext_vector_type(8)))  _Float16 v8h;
typedef __attribute__((ext_vector_type(8)))  float    v8f;

#define B_ 4
#define N_ 8192

__device__ __forceinline__ v16h cat8(v8h lo, v8h hi) {
  return __builtin_shufflevector(lo, hi, 0, 1, 2, 3, 4, 5, 6, 7,
                                 8, 9, 10, 11, 12, 13, 14, 15);
}
__device__ __forceinline__ v8h zero8() {
  v8h z = {(_Float16)0.f, (_Float16)0.f, (_Float16)0.f, (_Float16)0.f,
           (_Float16)0.f, (_Float16)0.f, (_Float16)0.f, (_Float16)0.f};
  return z;
}

// CDNA5 16-bit A-matrix 16x32 K index for v16h element e on this lane
// (used only by the pv1-conv1 gather path).
__device__ __forceinline__ int a_kmap(int lane, int e) {
  return ((e < 8) ? e : (e + 8)) + ((lane & 16) ? 8 : 0);
}

// Issue (no wait) an async DMA of a 16x32-half A chunk (rows m0..m0+15,
// k0..k0+31, row stride Kp halves) into a 1KB LDS slice, row-major [16][32].
// 2 x 16B per lane = 2 outstanding ASYNCcnt entries for this wave.
__device__ __forceinline__ void stage_issue(const _Float16* __restrict__ Wh,
                                            int m0, int Kp, int k0,
                                            _Float16* sl, int lane) {
#pragma unroll
  for (int j = 0; j < 2; ++j) {
    int i = lane + 32 * j;                 // 64 x 16B chunks
    int rr = i >> 2, pp = i & 3;
    unsigned long long ga =
        (unsigned long long)(size_t)(Wh + (size_t)(m0 + rr) * Kp + k0) +
        (unsigned long long)(pp * 16);
    unsigned la = (unsigned)(size_t)sl + (unsigned)(i * 16);
    asm volatile("global_load_async_to_lds_b128 %0, %1, off"
                 :: "v"(la), "v"(ga) : "memory");
  }
}
__device__ __forceinline__ void async_wait2() {
  asm volatile("s_wait_asynccnt 2" ::: "memory");
}
__device__ __forceinline__ void async_wait0() {
  asm volatile("s_wait_asynccnt 0" ::: "memory");
}

// Read this lane's A operand from a staged [16][32] LDS chunk.
__device__ __forceinline__ v16h read_a(const _Float16* sl, int row, int klo) {
  return cat8(*(const v8h*)(sl + (row << 5) + 2 * klo),
              *(const v8h*)(sl + (row << 5) + 2 * klo + 8));
}

// ---------------------------------------------------------------------------
__global__ void k_mean(const float* __restrict__ in, float* __restrict__ mean) {
  __shared__ float red[256];
  int b = blockIdx.x / 3, d = blockIdx.x % 3;
  const float* p = in + ((size_t)b * 6 + d) * N_;
  float s = 0.f;
  for (int i = threadIdx.x; i < N_; i += 256) s += p[i];
  red[threadIdx.x] = s;
  __syncthreads();
  for (int off = 128; off > 0; off >>= 1) {
    if (threadIdx.x < (unsigned)off) red[threadIdx.x] += red[threadIdx.x + off];
    __syncthreads();
  }
  if (threadIdx.x == 0) mean[b * 3 + d] = red[0] / (float)N_;
}

__global__ void k_ncidx(const float* __restrict__ in, const float* __restrict__ mean,
                        float* __restrict__ nc, int* __restrict__ idx, int r) {
  int t = blockIdx.x * blockDim.x + threadIdx.x;
  if (t >= B_ * N_) return;
  int b = t / N_, n = t % N_;
  int flat = 0;
  for (int d = 0; d < 3; ++d) {
    float c = in[((size_t)b * 6 + d) * N_ + n] - mean[b * 3 + d];
    float f = (c + 1.0f) * 0.5f * (float)r;
    f = fminf(fmaxf(f, 0.0f), (float)r - 1.0f);
    nc[((size_t)b * 3 + d) * N_ + n] = f;
    flat = flat * r + (int)rintf(f);
  }
  idx[t] = flat;
}

__global__ void k_fill(float* __restrict__ p, float v, size_t n) {
  size_t t = (size_t)blockIdx.x * blockDim.x + threadIdx.x;
  if (t < n) p[t] = v;
}

__global__ void k_scatter(const float* __restrict__ F, const int* __restrict__ idx,
                          float* __restrict__ gsum, float* __restrict__ cnt,
                          int C, int r3) {
  size_t t = (size_t)blockIdx.x * blockDim.x + threadIdx.x;
  size_t total = (size_t)B_ * C * N_;
  if (t >= total) return;
  int n = (int)(t % N_);
  int c = (int)((t / N_) % C);
  int b = (int)(t / ((size_t)N_ * C));
  int flat = idx[b * N_ + n];
  atomicAdd(&gsum[((size_t)b * C + c) * r3 + flat], F[((size_t)b * C + c) * N_ + n]);
  if (c == 0) atomicAdd(&cnt[(size_t)b * r3 + flat], 1.0f);
}

__global__ void k_avg(float* __restrict__ g, const float* __restrict__ cnt,
                      int C, int r3) {
  size_t t = (size_t)blockIdx.x * blockDim.x + threadIdx.x;
  size_t total = (size_t)B_ * C * r3;
  if (t >= total) return;
  int v = (int)(t % r3);
  int b = (int)(t / ((size_t)C * r3));
  g[t] /= fmaxf(cnt[(size_t)b * r3 + v], 1.0f);
}

// ---------------------------------------------------------------------------
// Pre-pack kernels: f32 -> f16, GEMM-friendly layouts.
__global__ void k_pack_w(const float* __restrict__ w, _Float16* __restrict__ o,
                         int M, int K, int Kp) {
  size_t t = (size_t)blockIdx.x * blockDim.x + threadIdx.x;
  if (t >= (size_t)M * Kp) return;
  int k = (int)(t % Kp);
  int m = (int)(t / Kp);
  o[t] = (k < K) ? (_Float16)w[(size_t)m * K + k] : (_Float16)0.f;
}

// cw (co, ci, 27) -> o[m][kk*ci + c]  (K = ci*27, kernel-position-major)
__global__ void k_pack_wconv(const float* __restrict__ cw, _Float16* __restrict__ o,
                             int co, int ci) {
  int K = ci * 27;
  size_t t = (size_t)blockIdx.x * blockDim.x + threadIdx.x;
  if (t >= (size_t)co * K) return;
  int m = (int)(t / K);
  int rest = (int)(t % K);
  int kk = rest / ci;
  int c = rest % ci;
  o[t] = (_Float16)cw[((size_t)m * ci + c) * 27 + kk];
}

// X (B,K,N) f32 -> o (B,N,Kp) f16 (zero-padded K tail)
__global__ void k_x2h_t(const float* __restrict__ X, _Float16* __restrict__ o,
                        int K, int Kp) {
  size_t t = (size_t)blockIdx.x * blockDim.x + threadIdx.x;
  size_t total = (size_t)B_ * N_ * Kp;
  if (t >= total) return;
  int k = (int)(t % Kp);
  int n = (int)((t / Kp) % N_);
  int b = (int)(t / ((size_t)Kp * N_));
  o[t] = (k < K) ? (_Float16)X[((size_t)b * K + k) * N_ + n] : (_Float16)0.f;
}

// G (B,C,r3) f32 -> o (B,r3,C) f16 (channels-last)
__global__ void k_grid2h_t(const float* __restrict__ G, _Float16* __restrict__ o,
                           int C, int r3) {
  size_t t = (size_t)blockIdx.x * blockDim.x + threadIdx.x;
  size_t total = (size_t)B_ * r3 * C;
  if (t >= total) return;
  int c = (int)(t % C);
  int v = (int)((t / C) % r3);
  int b = (int)(t / ((size_t)C * r3));
  o[t] = (_Float16)G[((size_t)b * C + c) * r3 + v];
}

// ---------------------------------------------------------------------------
// pv1-conv1 only (ci=6): gather-style implicit GEMM (tiny FLOP share).
__global__ void k_conv_wmma(const float* __restrict__ G, const float* __restrict__ W,
                            const float* __restrict__ bias,
                            const float* __restrict__ bng, const float* __restrict__ bnb,
                            const float* __restrict__ bnm, const float* __restrict__ bnv,
                            float* __restrict__ Y,
                            int ci, int co, int r, float eps, float slope) {
  int r3 = r * r * r;
  int lane = threadIdx.x & 31;
  int wave = blockIdx.x * (blockDim.x >> 5) + (threadIdx.x >> 5);
  int mt = co >> 4;
  int ctb = r3 >> 4;
  int total = B_ * ctb * mt;
  if (wave >= total) return;
  int mi = wave % mt;
  int ct = wave / mt;
  int b  = ct / ctb;
  int vb = (ct % ctb) << 4;
  int K = ci * 27;
  int m = (mi << 4) + (lane & 15);
  int v = vb + (lane & 15);
  int lr = (r == 32) ? 5 : 4;
  int d0 = v >> (2 * lr), d1 = (v >> lr) & (r - 1), d2 = v & (r - 1);
  int kbh = (lane & 16) ? 16 : 0;
  const float* Gb = G + (size_t)b * ci * r3;
  v8f acc = {0.f, 0.f, 0.f, 0.f, 0.f, 0.f, 0.f, 0.f};
  for (int k0 = 0; k0 < K; k0 += 32) {
    v16h av, bv;
#pragma unroll
    for (int e = 0; e < 16; ++e) {
      int ka = k0 + a_kmap(lane, e);
      av[e] = (ka < K) ? (_Float16)W[(size_t)m * K + ka] : (_Float16)0.f;
      int kk = k0 + a_kmap(lane, e);
      float x = 0.f;
      if (kk < K) {
        int c = kk / 27;
        int q = kk - c * 27;
        int z0 = d0 + q / 9 - 1;
        int rm = q % 9;
        int z1 = d1 + rm / 3 - 1;
        int z2 = d2 + rm % 3 - 1;
        if ((unsigned)z0 < (unsigned)r && (unsigned)z1 < (unsigned)r &&
            (unsigned)z2 < (unsigned)r)
          x = Gb[(size_t)c * r3 + ((z0 * r + z1) * r + z2)];
      }
      bv[e] = (_Float16)x;
    }
    acc = __builtin_amdgcn_wmma_f32_16x16x32_f16(false, av, false, bv,
                                                 (short)0, acc, false, false);
  }
  int rb = (lane & 16) ? 8 : 0;
#pragma unroll
  for (int e = 0; e < 8; ++e) {
    int row = (mi << 4) + rb + e;
    float y = acc[e] + bias[row];
    y = (y - bnm[row]) * rsqrtf(bnv[row] + eps) * bng[row] + bnb[row];
    y = (y > 0.f) ? y : y * slope;
    Y[((size_t)b * co + row) * r3 + v] = y;
  }
}

// ---------------------------------------------------------------------------
// Fast conv (ci = 64/128): channels-last f16 grid, packed f16 weights in
// kk-major order, double-buffered async-to-LDS A staging, 16x32 wave tile
// (2 voxel-column tiles share each staged A chunk).
__global__ void k_conv2(const _Float16* __restrict__ GT, const _Float16* __restrict__ Wc,
                        const float* __restrict__ bias,
                        const float* __restrict__ bng, const float* __restrict__ bnb,
                        const float* __restrict__ bnm, const float* __restrict__ bnv,
                        float* __restrict__ Y,
                        int ci, int cish, int co, int r, float eps, float slope) {
  __shared__ __align__(64) _Float16 aT[8 * 1024];
  int r3 = r * r * r;
  int lane = threadIdx.x & 31;
  int wv = threadIdx.x >> 5;
  int wave = blockIdx.x * 8 + wv;
  int mt = co >> 4;
  int ctb = r3 >> 5;               // 32 voxels per wave
  int total = B_ * ctb * mt;
  if (wave >= total) return;
  int mi = wave % mt;
  int ct = wave / mt;
  int b  = ct / ctb;
  int vb = (ct % ctb) << 5;
  int K = ci * 27;
  int m0 = mi << 4;
  int lr = (r == 32) ? 5 : 4;
  int v0 = vb + (lane & 15), v1 = v0 + 16;
  int d00 = v0 >> (2 * lr), d01 = (v0 >> lr) & (r - 1), d02 = v0 & (r - 1);
  int d10 = v1 >> (2 * lr), d11 = (v1 >> lr) & (r - 1), d12 = v1 & (r - 1);
  int kbh = (lane & 16) ? 16 : 0;
  int klo = (lane & 16) ? 8 : 0;
  int row = lane & 15;
  _Float16* sl0 = aT + wv * 1024;
  _Float16* sl1 = sl0 + 512;
  const _Float16* GTb = GT + (size_t)b * r3 * ci;
  v8f a0 = {0.f, 0.f, 0.f, 0.f, 0.f, 0.f, 0.f, 0.f};
  v8f a1 = a0;
  stage_issue(Wc, m0, K, 0, sl0, lane);
  for (int k0 = 0; k0 < K; k0 += 32) {
    _Float16* cur = (k0 & 32) ? sl1 : sl0;
    _Float16* nxt = (k0 & 32) ? sl0 : sl1;
    if (k0 + 32 < K) { stage_issue(Wc, m0, K, k0 + 32, nxt, lane); async_wait2(); }
    else             { async_wait0(); }
    v16h av = read_a(cur, row, klo);
    int start = k0 + kbh;            // 16-aligned; run stays within one kk
    int kk = start >> cish;
    int c0 = start & (ci - 1);
    int dz = kk / 9 - 1;
    int rm = kk % 9;
    int dy = rm / 3 - 1;
    int dx = rm % 3 - 1;
    int z0, z1, z2;
    v16h bv0, bv1;
    z0 = d00 + dz; z1 = d01 + dy; z2 = d02 + dx;
    if ((unsigned)z0 < (unsigned)r && (unsigned)z1 < (unsigned)r &&
        (unsigned)z2 < (unsigned)r) {
      const _Float16* gp = GTb + (size_t)((z0 * r + z1) * r + z2) * ci + c0;
      bv0 = cat8(*(const v8h*)gp, *(const v8h*)(gp + 8));
    } else {
      bv0 = cat8(zero8(), zero8());
    }
    z0 = d10 + dz; z1 = d11 + dy; z2 = d12 + dx;
    if ((unsigned)z0 < (unsigned)r && (unsigned)z1 < (unsigned)r &&
        (unsigned)z2 < (unsigned)r) {
      const _Float16* gp = GTb + (size_t)((z0 * r + z1) * r + z2) * ci + c0;
      bv1 = cat8(*(const v8h*)gp, *(const v8h*)(gp + 8));
    } else {
      bv1 = cat8(zero8(), zero8());
    }
    a0 = __builtin_amdgcn_wmma_f32_16x16x32_f16(false, av, false, bv0, (short)0, a0, false, false);
    a1 = __builtin_amdgcn_wmma_f32_16x16x32_f16(false, av, false, bv1, (short)0, a1, false, false);
  }
  int rb = (lane & 16) ? 8 : 0;
  v8f accs[2] = {a0, a1};
#pragma unroll
  for (int j = 0; j < 2; ++j) {
    int v = vb + (j << 4) + (lane & 15);
#pragma unroll
    for (int e = 0; e < 8; ++e) {
      int orow = m0 + rb + e;
      float y = accs[j][e] + bias[orow];
      y = (y - bnm[orow]) * rsqrtf(bnv[orow] + eps) * bng[orow] + bnb[orow];
      y = (y > 0.f) ? y : y * slope;
      Y[((size_t)b * co + orow) * r3 + v] = y;
    }
  }
}

// ---------------------------------------------------------------------------
// Pointwise GEMM: XT (B,N,Kp) f16, Wh (M,Kp) f16.  Wave tile = 16 x 64
// (4 accumulators share one double-buffered async-staged A chunk).
__global__ void k_gemm2(const _Float16* __restrict__ XT, const _Float16* __restrict__ Wh,
                        const float* __restrict__ bias,
                        const float* __restrict__ bng, const float* __restrict__ bnb,
                        const float* __restrict__ bnm, const float* __restrict__ bnv,
                        float* __restrict__ Y, int Kp, int M, float eps) {
  __shared__ __align__(64) _Float16 aT[8 * 1024];
  int lane = threadIdx.x & 31;
  int wv = threadIdx.x >> 5;
  int wave = blockIdx.x * 8 + wv;
  int mt = M >> 4;
  int ng = N_ >> 6;
  int total = B_ * ng * mt;
  if (wave >= total) return;
  int mi = wave % mt;
  int rest = wave / mt;
  int b = rest / ng;
  int nb = (rest % ng) << 6;
  int m0 = mi << 4;
  int row = lane & 15;
  int klo = (lane & 16) ? 8 : 0;
  int kbh = (lane & 16) ? 16 : 0;
  _Float16* sl0 = aT + wv * 1024;
  _Float16* sl1 = sl0 + 512;
  const _Float16* xb = XT + ((size_t)b * N_ + nb + row) * Kp;
  size_t cstride = (size_t)16 * Kp;
  v8f a0 = {0.f, 0.f, 0.f, 0.f, 0.f, 0.f, 0.f, 0.f};
  v8f a1 = a0, a2 = a0, a3 = a0;
  stage_issue(Wh, m0, Kp, 0, sl0, lane);
  for (int k0 = 0; k0 < Kp; k0 += 32) {
    _Float16* cur = (k0 & 32) ? sl1 : sl0;
    _Float16* nxt = (k0 & 32) ? sl0 : sl1;
    if (k0 + 32 < Kp) { stage_issue(Wh, m0, Kp, k0 + 32, nxt, lane); async_wait2(); }
    else              { async_wait0(); }
    v16h av = read_a(cur, row, klo);
    const _Float16* xk = xb + k0 + kbh;
    v16h b0 = cat8(*(const v8h*)(xk), *(const v8h*)(xk + 8));
    v16h b1 = cat8(*(const v8h*)(xk + cstride), *(const v8h*)(xk + cstride + 8));
    v16h b2 = cat8(*(const v8h*)(xk + 2 * cstride), *(const v8h*)(xk + 2 * cstride + 8));
    v16h b3 = cat8(*(const v8h*)(xk + 3 * cstride), *(const v8h*)(xk + 3 * cstride + 8));
    a0 = __builtin_amdgcn_wmma_f32_16x16x32_f16(false, av, false, b0, (short)0, a0, false, false);
    a1 = __builtin_amdgcn_wmma_f32_16x16x32_f16(false, av, false, b1, (short)0, a1, false, false);
    a2 = __builtin_amdgcn_wmma_f32_16x16x32_f16(false, av, false, b2, (short)0, a2, false, false);
    a3 = __builtin_amdgcn_wmma_f32_16x16x32_f16(false, av, false, b3, (short)0, a3, false, false);
  }
  int rb = (lane & 16) ? 8 : 0;
  v8f accs[4] = {a0, a1, a2, a3};
#pragma unroll
  for (int j = 0; j < 4; ++j) {
    int ncol = nb + (j << 4) + (lane & 15);
#pragma unroll
    for (int e = 0; e < 8; ++e) {
      int orow = m0 + rb + e;
      float y = accs[j][e] + bias[orow];
      y = (y - bnm[orow]) * rsqrtf(bnv[orow] + eps) * bng[orow] + bnb[orow];
      y = fmaxf(y, 0.f);
      Y[((size_t)b * M + orow) * N_ + ncol] = y;
    }
  }
}

// ---------------------------------------------------------------------------
// mlp5 (Kp=512 -> M=2048) + BN + relu fused with max-pool over N.
__global__ void k_mlp5_pool(const _Float16* __restrict__ XT, const _Float16* __restrict__ Wh,
                            const float* __restrict__ bias,
                            const float* __restrict__ bng, const float* __restrict__ bnb,
                            const float* __restrict__ bnm, const float* __restrict__ bnv,
                            float* __restrict__ pool, int Kp, int M) {
  __shared__ __align__(64) _Float16 aT[8 * 1024];
  int lane = threadIdx.x & 31;
  int wv = threadIdx.x >> 5;
  int wave = blockIdx.x * 8 + wv;
  int mt = M >> 4;
  int ng = N_ >> 6;
  int total = B_ * ng * mt;
  if (wave >= total) return;
  int mi = wave % mt;
  int rest = wave / mt;
  int b = rest / ng;
  int nb = (rest % ng) << 6;
  int m0 = mi << 4;
  int row = lane & 15;
  int klo = (lane & 16) ? 8 : 0;
  int kbh = (lane & 16) ? 16 : 0;
  _Float16* sl0 = aT + wv * 1024;
  _Float16* sl1 = sl0 + 512;
  const _Float16* xb = XT + ((size_t)b * N_ + nb + row) * Kp;
  size_t cstride = (size_t)16 * Kp;
  v8f a0 = {0.f, 0.f, 0.f, 0.f, 0.f, 0.f, 0.f, 0.f};
  v8f a1 = a0, a2 = a0, a3 = a0;
  stage_issue(Wh, m0, Kp, 0, sl0, lane);
  for (int k0 = 0; k0 < Kp; k0 += 32) {
    _Float16* cur = (k0 & 32) ? sl1 : sl0;
    _Float16* nxt = (k0 & 32) ? sl0 : sl1;
    if (k0 + 32 < Kp) { stage_issue(Wh, m0, Kp, k0 + 32, nxt, lane); async_wait2(); }
    else              { async_wait0(); }
    v16h av = read_a(cur, row, klo);
    const _Float16* xk = xb + k0 + kbh;
    v16h b0 = cat8(*(const v8h*)(xk), *(const v8h*)(xk + 8));
    v16h b1 = cat8(*(const v8h*)(xk + cstride), *(const v8h*)(xk + cstride + 8));
    v16h b2 = cat8(*(const v8h*)(xk + 2 * cstride), *(const v8h*)(xk + 2 * cstride + 8));
    v16h b3 = cat8(*(const v8h*)(xk + 3 * cstride), *(const v8h*)(xk + 3 * cstride + 8));
    a0 = __builtin_amdgcn_wmma_f32_16x16x32_f16(false, av, false, b0, (short)0, a0, false, false);
    a1 = __builtin_amdgcn_wmma_f32_16x16x32_f16(false, av, false, b1, (short)0, a1, false, false);
    a2 = __builtin_amdgcn_wmma_f32_16x16x32_f16(false, av, false, b2, (short)0, a2, false, false);
    a3 = __builtin_amdgcn_wmma_f32_16x16x32_f16(false, av, false, b3, (short)0, a3, false, false);
  }
  int rb = (lane & 16) ? 8 : 0;
  unsigned int* poolu = (unsigned int*)pool;
  v8f accs[4] = {a0, a1, a2, a3};
#pragma unroll
  for (int j = 0; j < 4; ++j) {
#pragma unroll
    for (int e = 0; e < 8; ++e) {
      int orow = m0 + rb + e;
      float y = accs[j][e] + bias[orow];
      y = (y - bnm[orow]) * rsqrtf(bnv[orow] + 1e-5f) * bng[orow] + bnb[orow];
      y = fmaxf(y, 0.f);
      y = fmaxf(y, __shfl_xor(y, 1, 16));
      y = fmaxf(y, __shfl_xor(y, 2, 16));
      y = fmaxf(y, __shfl_xor(y, 4, 16));
      y = fmaxf(y, __shfl_xor(y, 8, 16));
      if ((lane & 15) == 0)
        atomicMax(&poolu[(size_t)b * M + orow], __float_as_uint(y));
    }
  }
}

// ---------------------------------------------------------------------------
__global__ void k_se_pool(const float* __restrict__ g, float* __restrict__ s, int r3) {
  __shared__ float red[256];
  const float* p = g + (size_t)blockIdx.x * r3;
  float acc = 0.f;
  for (int i = threadIdx.x; i < r3; i += 256) acc += p[i];
  red[threadIdx.x] = acc;
  __syncthreads();
  for (int off = 128; off > 0; off >>= 1) {
    if (threadIdx.x < (unsigned)off) red[threadIdx.x] += red[threadIdx.x + off];
    __syncthreads();
  }
  if (threadIdx.x == 0) s[blockIdx.x] = red[0] / (float)r3;
}

__global__ void k_se_mlp(const float* __restrict__ s,
                         const float* __restrict__ w1, const float* __restrict__ b1,
                         const float* __restrict__ w2, const float* __restrict__ b2,
                         float* __restrict__ s2, int C, int red_) {
  __shared__ float ls[128];
  __shared__ float l1[32];
  int b = blockIdx.x, tid = threadIdx.x;
  if (tid < C) ls[tid] = s[b * C + tid];
  __syncthreads();
  if (tid < red_) {
    float a = b1[tid];
    for (int c = 0; c < C; ++c) a += w1[tid * C + c] * ls[c];
    l1[tid] = fmaxf(a, 0.f);
  }
  __syncthreads();
  if (tid < C) {
    float a = b2[tid];
    for (int j = 0; j < red_; ++j) a += w2[tid * red_ + j] * l1[j];
    s2[b * C + tid] = 1.0f / (1.0f + expf(-a));
  }
}

__global__ void k_se_scale(float* __restrict__ g, const float* __restrict__ s2, int r3) {
  size_t t = (size_t)blockIdx.x * blockDim.x + threadIdx.x;
  g[t] *= s2[t / (size_t)r3];
}

// ---------------------------------------------------------------------------
__global__ void k_devox_add(const float* __restrict__ g, const float* __restrict__ nc,
                            float* __restrict__ out, int C, int r) {
  int r3 = r * r * r;
  size_t t = (size_t)blockIdx.x * blockDim.x + threadIdx.x;
  size_t total = (size_t)B_ * C * N_;
  if (t >= total) return;
  int n = (int)(t % N_);
  int c = (int)((t / N_) % C);
  int b = (int)(t / ((size_t)N_ * C));
  float fx = nc[((size_t)b * 3 + 0) * N_ + n];
  float fy = nc[((size_t)b * 3 + 1) * N_ + n];
  float fz = nc[((size_t)b * 3 + 2) * N_ + n];
  int lx = (int)floorf(fx), ly = (int)floorf(fy), lz = (int)floorf(fz);
  float wx1 = fx - (float)lx, wy1 = fy - (float)ly, wz1 = fz - (float)lz;
  float wx0 = 1.f - wx1, wy0 = 1.f - wy1, wz0 = 1.f - wz1;
  int hx = min(lx + 1, r - 1), hy = min(ly + 1, r - 1), hz = min(lz + 1, r - 1);
  const float* gb = g + ((size_t)b * C + c) * r3;
#define GV(ix, iy, iz) gb[((ix) * r + (iy)) * r + (iz)]
  float val = GV(lx, ly, lz) * wx0 * wy0 * wz0 + GV(lx, ly, hz) * wx0 * wy0 * wz1 +
              GV(lx, hy, lz) * wx0 * wy1 * wz0 + GV(lx, hy, hz) * wx0 * wy1 * wz1 +
              GV(hx, ly, lz) * wx1 * wy0 * wz0 + GV(hx, ly, hz) * wx1 * wy0 * wz1 +
              GV(hx, hy, lz) * wx1 * wy1 * wz0 + GV(hx, hy, hz) * wx1 * wy1 * wz1;
#undef GV
  out[((size_t)b * C + c) * N_ + n] += val;
}

// ---------------------------------------------------------------------------
__global__ void k_classifier(const float* __restrict__ pool,
                             const float* __restrict__ w1, const float* __restrict__ b1,
                             const float* __restrict__ g1, const float* __restrict__ be1,
                             const float* __restrict__ m1, const float* __restrict__ v1,
                             const float* __restrict__ w2, const float* __restrict__ b2,
                             const float* __restrict__ g2, const float* __restrict__ be2,
                             const float* __restrict__ m2, const float* __restrict__ v2,
                             const float* __restrict__ w3, const float* __restrict__ b3,
                             float* __restrict__ out) {
  __shared__ float xs[2048];
  __shared__ float h1[512];
  __shared__ float h2[256];
  int b = blockIdx.x, tid = threadIdx.x;
  for (int i = tid; i < 2048; i += 256) xs[i] = pool[b * 2048 + i];
  __syncthreads();
  for (int o = tid; o < 512; o += 256) {
    float a = b1[o];
    for (int k = 0; k < 2048; ++k) a += w1[(size_t)o * 2048 + k] * xs[k];
    a = (a - m1[o]) * rsqrtf(v1[o] + 1e-5f) * g1[o] + be1[o];
    h1[o] = (a > 0.f) ? a : 0.2f * a;
  }
  __syncthreads();
  if (tid < 256) {
    float a = b2[tid];
    for (int k = 0; k < 512; ++k) a += w2[(size_t)tid * 512 + k] * h1[k];
    a = (a - m2[tid]) * rsqrtf(v2[tid] + 1e-5f) * g2[tid] + be2[tid];
    h2[tid] = (a > 0.f) ? a : 0.2f * a;
  }
  __syncthreads();
  if (tid < 40) {
    float a = b3[tid];
    for (int k = 0; k < 256; ++k) a += w3[tid * 256 + k] * h2[k];
    out[b * 40 + tid] = a;
  }
}

// ---------------------------------------------------------------------------
static inline int cdiv(long a, long b) { return (int)((a + b - 1) / b); }

extern "C" void kernel_launch(void* const* d_in, const int* in_sizes, int n_in,
                              void* d_out, int out_size, void* d_ws, size_t ws_size,
                              hipStream_t stream) {
  (void)in_sizes; (void)n_in; (void)out_size; (void)ws_size;
  const float* IN = (const float*)d_in[0];
  auto P = [&](int i) { return (const float*)d_in[i]; };

  float* ws = (float*)d_ws;
  // f32 workspace layout (float offsets).
  const size_t O_NC   = 16;                       // B*3*N = 98304
  const size_t O_IDX  = O_NC + 98304;             // B*N   = 32768 (ints)
  const size_t O_CNT  = O_IDX + 32768;            // B*32^3 = 131072
  const size_t O_GIN  = O_CNT + 131072;           // 2,097,152
  const size_t O_GA   = O_GIN + 2097152;          // 8,388,608
  const size_t O_GB   = O_GA + 8388608;           // 8,388,608
  const size_t O_SE1  = O_GB + 8388608;           // 512
  const size_t O_SE2  = O_SE1 + 512;              // 512
  const size_t O_FA   = O_SE2 + 512;              // 4,194,304
  const size_t O_FB   = O_FA + 4194304;           // 4,194,304
  const size_t O_F4   = O_FB + 4194304;           // 16,777,216
  const size_t O_POOL = O_F4 + 16777216;          // 8,192
  const size_t O_H    = O_POOL + 8192;            // f16 region below

  float* MEAN = ws;
  float* NC   = ws + O_NC;
  int*   IDX  = (int*)(ws + O_IDX);
  float* CNT  = ws + O_CNT;
  float* GIN  = ws + O_GIN;
  float* GA   = ws + O_GA;
  float* GB   = ws + O_GB;
  float* SE1  = ws + O_SE1;
  float* SE2  = ws + O_SE2;
  float* FA   = ws + O_FA;
  float* FB   = ws + O_FB;
  float* F4   = ws + O_F4;
  float* POOL = ws + O_POOL;

  _Float16* H = (_Float16*)(ws + O_H);
  // f16 workspace (half offsets).
  _Float16* HW_M1 = H + 0;          // 64  x 32   = 2048
  _Float16* HW_M2 = H + 2048;       // 128 x 64   = 8192
  _Float16* HW_M3 = H + 10240;      // 128 x 128  = 16384
  _Float16* HW_4  = H + 26624;      // 512 x 128  = 65536
  _Float16* HW_5  = H + 92160;      // 2048x 512  = 1048576
  _Float16* HC12  = H + 1140736;    // 64 x 1728  = 110592
  _Float16* HC21  = H + 1251328;    // 128x 1728  = 221184
  _Float16* HC22  = H + 1472512;    // 128x 3456  = 442368
  _Float16* HC31  = H + 1914880;    // 442368
  _Float16* HC32  = H + 2357248;    // 442368
  _Float16* HXT   = H + 2799616;    // B*N*512    = 16777216
  _Float16* HGT   = H + 19576832;   // B*32768*64 = 8388608

  k_mean<<<12, 256, 0, stream>>>(IN, MEAN);

  // Pack all GEMM weights to f16 (once per launch, deterministic).
  k_pack_w<<<cdiv(64L * 32, 256), 256, 0, stream>>>(P(27 + 17), HW_M1, 64, 6, 32);
  k_pack_w<<<cdiv(128L * 64, 256), 256, 0, stream>>>(P(49 + 17), HW_M2, 128, 64, 64);
  k_pack_w<<<cdiv(128L * 128, 256), 256, 0, stream>>>(P(71 + 17), HW_M3, 128, 128, 128);
  k_pack_w<<<cdiv(512L * 128, 256), 256, 0, stream>>>(P(20), HW_4, 512, 128, 128);
  k_pack_w<<<cdiv(2048L * 512, 256), 256, 0, stream>>>(P(26), HW_5, 2048, 512, 512);
  k_pack_wconv<<<cdiv(64L * 1728, 256), 256, 0, stream>>>(P(27 + 11), HC12, 64, 64);
  k_pack_wconv<<<cdiv(128L * 1728, 256), 256, 0, stream>>>(P(49 + 10), HC21, 128, 64);
  k_pack_wconv<<<cdiv(128L * 3456, 256), 256, 0, stream>>>(P(49 + 11), HC22, 128, 128);
  k_pack_wconv<<<cdiv(128L * 3456, 256), 256, 0, stream>>>(P(71 + 10), HC31, 128, 128);
  k_pack_wconv<<<cdiv(128L * 3456, 256), 256, 0, stream>>>(P(71 + 11), HC32, 128, 128);

  auto pv = [&](const float* F, float* Fout, int ci, int co, int r, int base,
                int red, const _Float16* Wc1, const _Float16* Wc2,
                const _Float16* Wm, int Kp) {
    int r3 = r * r * r;
    int cish = (ci == 64) ? 6 : 7;
    int cosh = (co == 64) ? 6 : 7;
    k_ncidx<<<cdiv((long)B_ * N_, 256), 256, 0, stream>>>(IN, MEAN, NC, IDX, r);
    k_fill<<<cdiv((long)B_ * ci * r3, 256), 256, 0, stream>>>(GIN, 0.f, (size_t)B_ * ci * r3);
    k_fill<<<cdiv((long)B_ * r3, 256), 256, 0, stream>>>(CNT, 0.f, (size_t)B_ * r3);
    k_scatter<<<cdiv((long)B_ * ci * N_, 256), 256, 0, stream>>>(F, IDX, GIN, CNT, ci, r3);
    k_avg<<<cdiv((long)B_ * ci * r3, 256), 256, 0, stream>>>(GIN, CNT, ci, r3);
    if (Wc1 == nullptr) {
      // pv1 conv1 (ci = 6): gather path, 16-voxel tiles
      int tiles = B_ * (r3 / 16) * (co / 16);
      k_conv_wmma<<<cdiv(tiles, 8), 256, 0, stream>>>(
          GIN, P(base + 10), P(base + 8),
          P(base + 1), P(base + 0), P(base + 2), P(base + 3),
          GA, ci, co, r, 1e-4f, 0.1f);
    } else {
      int tiles2 = B_ * (r3 / 32) * (co / 16);
      k_grid2h_t<<<cdiv((long)B_ * r3 * ci, 256), 256, 0, stream>>>(GIN, HGT, ci, r3);
      k_conv2<<<cdiv(tiles2, 8), 256, 0, stream>>>(
          HGT, Wc1, P(base + 8),
          P(base + 1), P(base + 0), P(base + 2), P(base + 3),
          GA, ci, cish, co, r, 1e-4f, 0.1f);
    }
    int tiles2 = B_ * (r3 / 32) * (co / 16);
    k_grid2h_t<<<cdiv((long)B_ * r3 * co, 256), 256, 0, stream>>>(GA, HGT, co, r3);
    k_conv2<<<cdiv(tiles2, 8), 256, 0, stream>>>(
        HGT, Wc2, P(base + 9),
        P(base + 5), P(base + 4), P(base + 6), P(base + 7),
        GB, co, cosh, co, r, 1e-4f, 0.1f);
    k_se_pool<<<B_ * co, 256, 0, stream>>>(GB, SE1, r3);
    k_se_mlp<<<B_, 128, 0, stream>>>(SE1, P(base + 20), P(base + 18),
                                     P(base + 21), P(base + 19), SE2, co, red);
    k_se_scale<<<cdiv((long)B_ * co * r3, 256), 256, 0, stream>>>(GB, SE2, r3);
    // shared MLP via f16 GEMM
    k_x2h_t<<<cdiv((long)B_ * N_ * Kp, 256), 256, 0, stream>>>(F, HXT, ci, Kp);
    int gtiles = B_ * (N_ / 64) * (co / 16);
    k_gemm2<<<cdiv(gtiles, 8), 256, 0, stream>>>(
        HXT, Wm, P(base + 12),
        P(base + 14), P(base + 13), P(base + 15), P(base + 16),
        Fout, Kp, co, 1e-5f);
    k_devox_add<<<cdiv((long)B_ * co * N_, 256), 256, 0, stream>>>(GB, NC, Fout, co, r);
  };

  pv(IN, FA, 6, 64, 32, 27, 8, nullptr, HC12, HW_M1, 32);
  pv(FA, FB, 64, 128, 16, 49, 16, HC21, HC22, HW_M2, 64);
  pv(FB, FA, 128, 128, 16, 71, 16, HC31, HC32, HW_M3, 128);

  // mlp4: 128 -> 512
  k_x2h_t<<<cdiv((long)B_ * N_ * 128, 256), 256, 0, stream>>>(FA, HXT, 128, 128);
  {
    int gtiles = B_ * (N_ / 64) * (512 / 16);
    k_gemm2<<<cdiv(gtiles, 8), 256, 0, stream>>>(
        HXT, HW_4, P(15), P(17), P(16), P(18), P(19), F4, 128, 512, 1e-5f);
  }

  // mlp5 (512 -> 2048) fused with max-pool over N.
  k_x2h_t<<<cdiv((long)B_ * N_ * 512, 256), 256, 0, stream>>>(F4, HXT, 512, 512);
  k_fill<<<cdiv((long)B_ * 2048, 256), 256, 0, stream>>>(POOL, 0.f, (size_t)B_ * 2048);
  {
    int gtiles = B_ * (N_ / 64) * (2048 / 16);
    k_mlp5_pool<<<cdiv(gtiles, 8), 256, 0, stream>>>(
        HXT, HW_5, P(21), P(23), P(22), P(24), P(25), POOL, 512, 2048);
  }

  // Classifier head.
  k_classifier<<<B_, 256, 0, stream>>>(
      POOL, P(12), P(1), P(5), P(4), P(6), P(7),
      P(13), P(2), P(9), P(8), P(10), P(11),
      P(14), P(3), (float*)d_out);
}
